// Attention_6468220748045
// MI455X (gfx1250) — matrix operand order repaired and checked by reading, not compiled, not run on hardware
//
#include <hip/hip_runtime.h>
#include <hip/hip_bf16.h>

// ---------------------------------------------------------------------------
// Problem constants (from reference): B=128, S=179, E=1024, H=16, D=64
// ---------------------------------------------------------------------------
constexpr int S_SEQ  = 179;
constexpr int E_DIM  = 1024;
constexpr int H_HEAD = 16;
constexpr int D_HEAD = 64;
constexpr int M_TOT  = 128 * S_SEQ;      // 22912 rows, divisible by 128
constexpr int S_PAD  = 192;              // 179 padded to 12 WMMA tiles

typedef __attribute__((ext_vector_type(16))) __bf16 v16bf;
typedef __attribute__((ext_vector_type(8)))  float  v8f;

union FragBF {
  v16bf v;
  uint2 q[4];
};

__device__ __forceinline__ unsigned short f2bf(float f) {
  union { float f; unsigned int u; } v; v.f = f;
  unsigned int u = v.u;
  unsigned int r = (u + 0x7FFFu + ((u >> 16) & 1u)) >> 16;  // RNE
  return (unsigned short)r;
}

__device__ __forceinline__ unsigned int f2bf_pk(float a, float b) {
#if __has_builtin(__builtin_amdgcn_cvt_pk_bf16_f32)
  auto r = __builtin_amdgcn_cvt_pk_bf16_f32(a, b);
  union { decltype(r) v; unsigned int u; } cv; cv.v = r;
  return cv.u;
#else
  return (unsigned int)f2bf(a) | ((unsigned int)f2bf(b) << 16);
#endif
}

// A fragment (16x32 bf16, M x K), documented layout:
// lane L: M = L&15; element j: K = (j<8?0:16) + (L>>4)*8 + (j&7)
__device__ __forceinline__ v16bf load_frag_a(const unsigned short* p) {
  FragBF f;
  f.q[0] = *(const uint2*)(p);
  f.q[1] = *(const uint2*)(p + 4);
  f.q[2] = *(const uint2*)(p + 16);
  f.q[3] = *(const uint2*)(p + 20);
  return f.v;
}

// B fragment (32x16 bf16, K x N): lane L: N = L&15; element j: K = (j<8?0:16) + (L>>4)*8 + (j&7)
// -> strided gather (stride ld) from an LDS tile stored [K][N] row-major.
__device__ __forceinline__ v16bf load_frag_b(const unsigned short* p, int ld) {
  union { v16bf v; unsigned short u[16]; } f;
  #pragma unroll
  for (int j = 0; j < 8; ++j) {
    f.u[j]     = p[j * ld];
    f.u[j + 8] = p[(j + 16) * ld];
  }
  return f.v;
}

// ---------------------------------------------------------------------------
// Prepass 1: x (f32, row-major [M][E]) -> xb (bf16, same layout)
// ---------------------------------------------------------------------------
__global__ __launch_bounds__(256)
void cvt_x_kernel(const float* __restrict__ x, unsigned short* __restrict__ xb) {
  size_t base = ((size_t)blockIdx.x * 256 + threadIdx.x) * 8;
  float4 f0 = *(const float4*)(x + base);
  float4 f1 = *(const float4*)(x + base + 4);
  uint4 o;
  o.x = f2bf_pk(f0.x, f0.y);
  o.y = f2bf_pk(f0.z, f0.w);
  o.z = f2bf_pk(f1.x, f1.y);
  o.w = f2bf_pk(f1.z, f1.w);
  *(uint4*)(xb + base) = o;
}

// ---------------------------------------------------------------------------
// Prepass 2: W (f32 [N][K]) -> Wt (bf16 [K][N]) for each of Wq/Wk/Wv.
// Tiny (12 MB read total); removes all transposition from the GEMM hot loop.
// ---------------------------------------------------------------------------
__global__ __launch_bounds__(256)
void cvt_w_kernel(const float* __restrict__ Wq, const float* __restrict__ Wk,
                  const float* __restrict__ Wv, unsigned short* __restrict__ Wt) {
  const int z = blockIdx.z;
  const float* W = (z == 0) ? Wq : (z == 1) ? Wk : Wv;
  unsigned short* dst = Wt + (size_t)z * E_DIM * E_DIM;
  int t = blockIdx.x * 256 + threadIdx.x;     // 0..131071
  int n  = t >> 7;
  int k8 = (t & 127) * 8;
  float4 f0 = *(const float4*)(W + (size_t)n * E_DIM + k8);
  float4 f1 = *(const float4*)(W + (size_t)n * E_DIM + k8 + 4);
  float v[8] = {f0.x, f0.y, f0.z, f0.w, f1.x, f1.y, f1.z, f1.w};
  #pragma unroll
  for (int j = 0; j < 8; ++j)
    dst[(size_t)(k8 + j) * E_DIM + n] = f2bf(v[j]);
}

// ---------------------------------------------------------------------------
// Kernel 1: QKV projection GEMM, all-bf16 inputs.
// Block: 128 threads = 4 waves; tile 128(M) x 64(N); K-step 32, double-buffered.
// Each wave: 32x64 strip = 2x4 WMMA tiles (8 WMMA per 6 fragment loads).
// Q,V written [B,H,S,D]; K written transposed+padded [B,H,D,192].
// ---------------------------------------------------------------------------
__global__ __launch_bounds__(128)
void qkv_proj_kernel(const unsigned short* __restrict__ xb,
                     const unsigned short* __restrict__ Wt,
                     unsigned short* __restrict__ Qo,
                     unsigned short* __restrict__ Ko,
                     unsigned short* __restrict__ Vo) {
  constexpr int LDA = 36;   // 32 + 4 pad (bf16 units)
  constexpr int LDB = 68;   // 64 + 4 pad
  constexpr int KT_STEPS = E_DIM / 32;
  const int nt0 = blockIdx.x * 64;
  const int mt0 = blockIdx.y * 128;
  const int z   = blockIdx.z;
  const unsigned short* Wz = Wt + (size_t)z * E_DIM * E_DIM;  // [K][N] bf16
  unsigned short* dst = (z == 0) ? Qo : (z == 1) ? Ko : Vo;

  __shared__ unsigned short Atile[2][128 * LDA];
  __shared__ unsigned short Btile[2][32 * LDB];

  const int tid  = threadIdx.x;
  const int lane = tid & 31;
  const int w    = tid >> 5;
  const int half = lane >> 4;
  const int wm   = w * 32;

  auto load_tiles = [&](int kt, int buf) {
    const int k0 = kt * 32;
    // A: xb[mt0+r][k0 + c8..c8+7] -> Atile ; 128 rows x 4 groups of 8
    #pragma unroll
    for (int i = 0; i < 4; ++i) {
      int lin = tid + i * 128;            // 0..511
      int r = lin >> 2, c8 = (lin & 3) * 8;
      uint4 v = *(const uint4*)(xb + (size_t)(mt0 + r) * E_DIM + k0 + c8);
      *(uint2*)(&Atile[buf][r * LDA + c8])     = make_uint2(v.x, v.y);
      *(uint2*)(&Atile[buf][r * LDA + c8 + 4]) = make_uint2(v.z, v.w);
    }
    // B: Wz[k0+k][nt0 + c8..c8+7] -> Btile ; 32 rows x 8 groups of 8
    #pragma unroll
    for (int i = 0; i < 2; ++i) {
      int lin = tid + i * 128;            // 0..255
      int k = lin >> 3, c8 = (lin & 7) * 8;
      uint4 v = *(const uint4*)(Wz + (size_t)(k0 + k) * E_DIM + nt0 + c8);
      *(uint2*)(&Btile[buf][k * LDB + c8])     = make_uint2(v.x, v.y);
      *(uint2*)(&Btile[buf][k * LDB + c8 + 4]) = make_uint2(v.z, v.w);
    }
  };

  v8f zero8 = {0.f, 0.f, 0.f, 0.f, 0.f, 0.f, 0.f, 0.f};
  v8f acc[2][4] = {{zero8, zero8, zero8, zero8}, {zero8, zero8, zero8, zero8}};

  load_tiles(0, 0);
  __syncthreads();

  for (int kt = 0; kt < KT_STEPS; ++kt) {
    int buf = kt & 1;
    if (kt + 1 < KT_STEPS) load_tiles(kt + 1, buf ^ 1);
    if (kt + 2 < KT_STEPS) {
      // L2 prefetch of the k+2 A tile (global_prefetch_b8)
      __builtin_prefetch(xb + (size_t)(mt0 + (tid >> 2)) * E_DIM + (kt + 2) * 32, 0, 0);
    }

    v16bf a[2], b[4];
    #pragma unroll
    for (int i = 0; i < 2; ++i)
      a[i] = load_frag_a(&Atile[buf][(wm + i * 16 + (lane & 15)) * LDA + half * 8]);
    #pragma unroll
    for (int j = 0; j < 4; ++j)
      b[j] = load_frag_b(&Btile[buf][(half * 8) * LDB + j * 16 + (lane & 15)], LDB);

    #pragma unroll
    for (int i = 0; i < 2; ++i)
      #pragma unroll
      for (int j = 0; j < 4; ++j)
        acc[i][j] = __builtin_amdgcn_wmma_f32_16x16x32_bf16(
            false, a[i], false, b[j], (short)0, acc[i][j], false, false);
    __syncthreads();
  }

  // C/D layout: VGPR r, lane L -> M = r + 8*(L>>4), N = L&15
  #pragma unroll
  for (int i = 0; i < 2; ++i)
    #pragma unroll
    for (int j = 0; j < 4; ++j)
      #pragma unroll
      for (int r = 0; r < 8; ++r) {
        int m = mt0 + wm + i * 16 + r + 8 * half;
        int n = nt0 + j * 16 + (lane & 15);
        int bb = m / S_SEQ, ss = m % S_SEQ;
        int hh = n >> 6,    dd = n & 63;
        size_t idx;
        if (z == 1)  // K transposed + padded: [B,H,D,S_PAD]
          idx = (((size_t)bb * H_HEAD + hh) * D_HEAD + dd) * S_PAD + ss;
        else         // Q,V: [B,H,S,D]
          idx = (((size_t)bb * H_HEAD + hh) * S_SEQ + ss) * D_HEAD + dd;
        dst[idx] = f2bf(acc[i][j][r]);
      }
}

// ---------------------------------------------------------------------------
// Kernel 2: attention for one (b,h) and a 32-row query chunk.
// scores = (Q K^T) * E^-0.5 * (|i-j|/179); softmax; O = P V.
// K arrives pre-transposed [d][192] so its LDS tile is a straight copy;
// padded key columns are masked to -1e30 before softmax.
// ---------------------------------------------------------------------------
__global__ __launch_bounds__(128)
void attn_kernel(const unsigned short* __restrict__ Q,
                 const unsigned short* __restrict__ Kt,
                 const unsigned short* __restrict__ V,
                 float* __restrict__ O) {
  constexpr int LQ = 68;    // 64+4
  constexpr int LK = 196;   // 192+4
  constexpr int LV = 68;
  constexpr int LS = 196;
  const int qc = blockIdx.x;          // 0..5
  const int bh = blockIdx.y;          // 0..B*H-1
  const int q0 = qc * 32;
  const int tid  = threadIdx.x;
  const int lane = tid & 31;
  const int w    = tid >> 5;
  const int half = lane >> 4;

  __shared__ union { unsigned short Qs[32 * LQ]; unsigned short Ps[32 * LS]; } u1;
  __shared__ union { unsigned short KT[D_HEAD * LK]; unsigned short Vs[S_PAD * LV]; } u2;
  __shared__ float Ss[32 * LS];

  const unsigned short* Qg = Q  + (size_t)bh * S_SEQ * D_HEAD;
  const unsigned short* Kg = Kt + (size_t)bh * D_HEAD * S_PAD;
  const unsigned short* Vg = V  + (size_t)bh * S_SEQ * D_HEAD;

  // --- stage 1 loads: Q chunk and pre-transposed K tile (plain copy) ---
  #pragma unroll
  for (int i = 0; i < 4; ++i) {
    int lin = tid + i * 128;            // 32 rows x 16 groups of 4
    int r = lin >> 4, c4 = (lin & 15) * 4;
    int s = q0 + r;
    uint2 v; v.x = 0u; v.y = 0u;
    if (s < S_SEQ) v = *(const uint2*)(Qg + (size_t)s * D_HEAD + c4);
    *(uint2*)(&u1.Qs[r * LQ + c4]) = v;
  }
  #pragma unroll
  for (int i = 0; i < 24; ++i) {
    int lin = tid + i * 128;            // 64 rows x 48 groups of 4
    int d = lin / 48, c4 = (lin % 48) * 4;
    *(uint2*)(&u2.KT[d * LK + c4]) = *(const uint2*)(Kg + (size_t)d * S_PAD + c4);
  }
  __syncthreads();

  // --- scores: 2x12 = 24 WMMA tiles over 4 waves ---
  for (int t = w; t < 24; t += 4) {
    int mt = t / 12, nt = t % 12;
    v8f acc = {0.f, 0.f, 0.f, 0.f, 0.f, 0.f, 0.f, 0.f};
    #pragma unroll
    for (int ks = 0; ks < 2; ++ks) {
      v16bf a = load_frag_a(&u1.Qs[(mt * 16 + (lane & 15)) * LQ + ks * 32 + half * 8]);
      v16bf b = load_frag_b(&u2.KT[(ks * 32 + half * 8) * LK + nt * 16 + (lane & 15)], LK);
      acc = __builtin_amdgcn_wmma_f32_16x16x32_bf16(
          false, a, false, b, (short)0, acc, false, false);
    }
    #pragma unroll
    for (int r = 0; r < 8; ++r) {
      int ml  = mt * 16 + r + 8 * half;
      int kgl = nt * 16 + (lane & 15);
      int qgl = q0 + ml;
      float bias = fabsf((float)(qgl - kgl)) * (1.0f / 179.0f);
      float val  = acc[r] * 0.03125f * bias;      // E^-0.5 = 1/32
      if (kgl >= S_SEQ) val = -1e30f;             // mask padded keys
      Ss[ml * LS + kgl] = val;
    }
  }
  __syncthreads();

  // --- stage 2: softmax (4 threads per row) + load V over KT's space ---
  #pragma unroll
  for (int i = 0; i < 24; ++i) {
    int lin = tid + i * 128;            // 192 rows x 16 groups of 4
    int s = lin >> 4, c4 = (lin & 15) * 4;
    uint2 v; v.x = 0u; v.y = 0u;
    if (s < S_SEQ) v = *(const uint2*)(Vg + (size_t)s * D_HEAD + c4);
    *(uint2*)(&u2.Vs[s * LV + c4]) = v;
  }
  {
    int row = tid >> 2, qq = tid & 3;
    float mx = -3.0e38f;
    for (int c = qq; c < S_PAD; c += 4) mx = fmaxf(mx, Ss[row * LS + c]);
    mx = fmaxf(mx, __shfl_xor(mx, 1));
    mx = fmaxf(mx, __shfl_xor(mx, 2));
    float sum = 0.f;
    for (int c = qq; c < S_PAD; c += 4) {
      float e = __expf(Ss[row * LS + c] - mx);
      Ss[row * LS + c] = e;
      sum += e;
    }
    sum += __shfl_xor(sum, 1);
    sum += __shfl_xor(sum, 2);
    float inv = 1.0f / sum;
    for (int c = qq; c < S_PAD; c += 4)
      u1.Ps[row * LS + c] = f2bf(Ss[row * LS + c] * inv);   // Qs dead here
  }
  __syncthreads();

  // --- O = P V : 2x4 = 8 WMMA tiles over 4 waves, K over 192 keys ---
  for (int t = w; t < 8; t += 4) {
    int mt = t >> 2, nt = t & 3;
    v8f acc = {0.f, 0.f, 0.f, 0.f, 0.f, 0.f, 0.f, 0.f};
    #pragma unroll
    for (int ks = 0; ks < 6; ++ks) {
      v16bf a = load_frag_a(&u1.Ps[(mt * 16 + (lane & 15)) * LS + ks * 32 + half * 8]);
      v16bf b = load_frag_b(&u2.Vs[(ks * 32 + half * 8) * LV + nt * 16 + (lane & 15)], LV);
      acc = __builtin_amdgcn_wmma_f32_16x16x32_bf16(
          false, a, false, b, (short)0, acc, false, false);
    }
    #pragma unroll
    for (int r = 0; r < 8; ++r) {
      int ml = mt * 16 + r + 8 * half;
      int sg = q0 + ml;
      if (sg < S_SEQ) {
        int d = nt * 16 + (lane & 15);
        O[((size_t)bh * S_SEQ + sg) * D_HEAD + d] = acc[r];
      }
    }
  }
}

// ---------------------------------------------------------------------------
// Kernel 3: LayerNorm over E=1024, fusing the [B,H,S,D]->[B,S,E] permute.
// ---------------------------------------------------------------------------
__global__ __launch_bounds__(256)
void ln_kernel(const float* __restrict__ O,
               const float* __restrict__ gamma,
               const float* __restrict__ beta,
               float* __restrict__ out) {
  const int bs = blockIdx.x;          // b*179 + s
  const int b = bs / S_SEQ, s = bs % S_SEQ;
  const int tid = threadIdx.x;

  float v[4];
  float sum = 0.f, sq = 0.f;
  #pragma unroll
  for (int i = 0; i < 4; ++i) {
    int e = tid + i * 256;
    int h = e >> 6, d = e & 63;
    float xv = O[(((size_t)b * H_HEAD + h) * S_SEQ + s) * D_HEAD + d];
    v[i] = xv; sum += xv; sq += xv * xv;
  }
  #pragma unroll
  for (int off = 16; off; off >>= 1) {
    sum += __shfl_down(sum, off);
    sq  += __shfl_down(sq, off);
  }
  __shared__ float ssum[8], ssq[8];
  int wv = tid >> 5, lane = tid & 31;
  if (lane == 0) { ssum[wv] = sum; ssq[wv] = sq; }
  __syncthreads();
  if (tid == 0) {
    float a = 0.f, c = 0.f;
    #pragma unroll
    for (int i = 0; i < 8; ++i) { a += ssum[i]; c += ssq[i]; }
    ssum[0] = a; ssq[0] = c;
  }
  __syncthreads();
  float mean = ssum[0] * (1.0f / 1024.0f);
  float var  = ssq[0]  * (1.0f / 1024.0f) - mean * mean;
  float rstd = rsqrtf(var + 1e-5f);
  #pragma unroll
  for (int i = 0; i < 4; ++i) {
    int e = tid + i * 256;
    out[(size_t)bs * E_DIM + e] = (v[i] - mean) * rstd * gamma[e] + beta[e];
  }
}

// ---------------------------------------------------------------------------
extern "C" void kernel_launch(void* const* d_in, const int* in_sizes, int n_in,
                              void* d_out, int out_size, void* d_ws, size_t ws_size,
                              hipStream_t stream) {
  (void)in_sizes; (void)n_in; (void)out_size; (void)ws_size;
  const float* x     = (const float*)d_in[0];
  const float* Wq    = (const float*)d_in[1];
  const float* Wk    = (const float*)d_in[2];
  const float* Wv    = (const float*)d_in[3];
  const float* gamma = (const float*)d_in[4];
  const float* beta  = (const float*)d_in[5];
  float* out = (float*)d_out;

  const size_t qElems = (size_t)M_TOT * E_DIM;                      // Q, V (bf16)
  const size_t kElems = (size_t)128 * H_HEAD * D_HEAD * S_PAD;      // K padded (bf16)
  unsigned short* Qb = (unsigned short*)d_ws;
  unsigned short* Kb = Qb + qElems;
  unsigned short* Vb = Kb + kElems;
  // Region after QKV: O (f32) for attn->LN, aliased with the bf16 staging
  // buffers (xb, Wt) whose lifetime ends before O is written.
  unsigned short* base2 = Vb + qElems;
  float* O = (float*)base2;
  unsigned short* xb = (unsigned short*)base2;        // bf16 x, [M][E]
  unsigned short* Wt = xb + qElems;                   // bf16 W^T x3, [K][N]

  cvt_x_kernel<<<dim3((unsigned)(qElems / 2048)), 256, 0, stream>>>(x, xb);
  cvt_w_kernel<<<dim3(512, 1, 3), 256, 0, stream>>>(Wq, Wk, Wv, Wt);
  qkv_proj_kernel<<<dim3(E_DIM / 64, M_TOT / 128, 3), 128, 0, stream>>>(
      xb, Wt, Qb, Kb, Vb);
  attn_kernel<<<dim3(S_PAD / 32, 128 * H_HEAD), 128, 0, stream>>>(Qb, Kb, Vb, O);
  ln_kernel<<<dim3(M_TOT), 256, 0, stream>>>(O, gamma, beta, out);
}
